// BitLinear_63204738728236
// MI455X (gfx1250) — compile-verified, matrix operand order checked by reading, not run on hardware
//
#include <hip/hip_runtime.h>
#include <stdint.h>

// ---------------------------------------------------------------------------
// BitLinear (ternary weights x int8 activations) for MI455X / gfx1250.
// GEMM: M=8192 (B*S), N=4096 (D_out), K=4096 (D_in) using
// v_wmma_i32_16x16x64_iu8 (exact integer math == reference einsum).
//
// The int8 intermediates are stored with the K-order inside every 64-wide
// K block permuted so that each lane's WMMA fragment (32 bytes) is fully
// contiguous -> 2x global_load_b128 per fragment.
//   natural k = seg*16 + half*8 + off  (seg 0..3, half 0..1, off 0..7)
//   stored  p = half*32 + seg*8 + off
//
// GEMM tiling: 64x64 wave tile (4x4 fragments) -> 16 b128 loads : 16 WMMAs
// per K-step, ~2.1 GB total L2 operand traffic (both int8 operands are
// L2-resident: 48 MB << 192 MB).
// ---------------------------------------------------------------------------

typedef __attribute__((ext_vector_type(8))) int v8i;

#define M_TOK   8192
#define N_OUT   4096
#define K_IN    4096
#define W_ELEMS (N_OUT * K_IN)          // 16,777,216
#define EPSF    1e-8f
#define QMAXF   127.0f

// ---------------- workspace layout (bytes) ----------------
#define WS_PARTIAL_OFF  256
#define WS_COMB_OFF     8192
#define WS_WQ_OFF       65536
#define WS_AQ_OFF       (65536 + (size_t)W_ELEMS)

// Permute a packed-dword index within its 16-dword (64-value) K block:
// dword l (= k/4, 4 int8 per dword): half=(l>>1)&1, seg=l>>2, offd=l&1
// stored dword position = half*8 + seg*2 + offd
__device__ __forceinline__ int shuffle_dw(int idx) {
    const int l = idx & 15;
    return (idx & ~15) | ((l & 2) << 2) | ((l >> 2) << 1) | (l & 1);
}

// ============================================================
// Kernel 1: partial sums of |W| (grid-stride, float4 loads)
// ============================================================
__global__ __launch_bounds__(256) void wabs_partial_kernel(
    const float4* __restrict__ w4, float* __restrict__ partial)
{
    float s = 0.0f;
    const int tid = blockIdx.x * 256 + threadIdx.x;
    const int nthreads = 1024 * 256;
    for (int i = tid; i < W_ELEMS / 4; i += nthreads) {
        float4 v = w4[i];
        s += fabsf(v.x) + fabsf(v.y) + fabsf(v.z) + fabsf(v.w);
    }
    __shared__ float sm[256];
    sm[threadIdx.x] = s;
    __syncthreads();
    for (int stride = 128; stride > 0; stride >>= 1) {
        if (threadIdx.x < stride) sm[threadIdx.x] += sm[threadIdx.x + stride];
        __syncthreads();
    }
    if (threadIdx.x == 0) partial[blockIdx.x] = sm[0];
}

// ============================================================
// Kernel 2: final reduce -> w_scale = mean(|W|) + eps
// ============================================================
__global__ __launch_bounds__(256) void wscale_final_kernel(
    const float* __restrict__ partial, float* __restrict__ wscale)
{
    __shared__ float sm[256];
    float s = partial[threadIdx.x] + partial[threadIdx.x + 256] +
              partial[threadIdx.x + 512] + partial[threadIdx.x + 768];
    sm[threadIdx.x] = s;
    __syncthreads();
    for (int stride = 128; stride > 0; stride >>= 1) {
        if (threadIdx.x < stride) sm[threadIdx.x] += sm[threadIdx.x + stride];
        __syncthreads();
    }
    if (threadIdx.x == 0) *wscale = sm[0] / (float)W_ELEMS + EPSF;
}

// ============================================================
// Kernel 3: ternary weight quant, 4 packed int8 per dword,
//           written in fragment-ready (K-permuted) layout
// ============================================================
__device__ __forceinline__ int tern1(float x, float inv) {
    float r = rintf(x * inv);                 // RNE, matches jnp.round
    r = fminf(fmaxf(r, -1.0f), 1.0f);
    return (int)r;
}

__global__ __launch_bounds__(256) void quant_w_kernel(
    const float4* __restrict__ w4, const float* __restrict__ wscale,
    int* __restrict__ wq_packed)
{
    const float inv = 1.0f / (*wscale);
    const int idx = blockIdx.x * 256 + threadIdx.x;   // one float4 -> one dword
    if (idx < W_ELEMS / 4) {
        float4 v = w4[idx];
        int q0 = tern1(v.x, inv), q1 = tern1(v.y, inv);
        int q2 = tern1(v.z, inv), q3 = tern1(v.w, inv);
        wq_packed[shuffle_dw(idx)] =
            (q0 & 0xFF) | ((q1 & 0xFF) << 8) |
            ((q2 & 0xFF) << 16) | ((q3 & 0xFF) << 24);
    }
}

// ============================================================
// Kernel 4: per-token activation quant + fused output scale
//   one 256-thread block per token row of 4096 floats
// ============================================================
__device__ __forceinline__ int q8(float x, float k) {
    float v = fminf(fmaxf(x * k, -128.0f), 127.0f);   // clip then round (ref order)
    return (int)rintf(v);
}

__global__ __launch_bounds__(256) void quant_a_kernel(
    const float* __restrict__ x, const float* __restrict__ wscale,
    int* __restrict__ aq_packed, float* __restrict__ comb)
{
    const int m = blockIdx.x;
    const int tid = threadIdx.x;
    const float4* row = (const float4*)(x + (size_t)m * K_IN);

    float4 v[4];
    float mx = 0.0f;
    #pragma unroll
    for (int j = 0; j < 4; ++j) {
        v[j] = row[tid + 256 * j];
        mx = fmaxf(mx, fmaxf(fmaxf(fabsf(v[j].x), fabsf(v[j].y)),
                             fmaxf(fabsf(v[j].z), fabsf(v[j].w))));
    }
    __shared__ float sm[256];
    sm[tid] = mx;
    __syncthreads();
    for (int stride = 128; stride > 0; stride >>= 1) {
        if (tid < stride) sm[tid] = fmaxf(sm[tid], sm[tid + stride]);
        __syncthreads();
    }
    const float a_scale = sm[0] + EPSF;
    if (tid == 0) comb[m] = (*wscale) * a_scale * (1.0f / QMAXF);

    const float k = QMAXF / a_scale;
    const int rowbase = m * (K_IN / 4);
    #pragma unroll
    for (int j = 0; j < 4; ++j) {
        int q0 = q8(v[j].x, k), q1 = q8(v[j].y, k);
        int q2 = q8(v[j].z, k), q3 = q8(v[j].w, k);
        aq_packed[shuffle_dw(rowbase + tid + 256 * j)] =
            (q0 & 0xFF) | ((q1 & 0xFF) << 8) |
            ((q2 & 0xFF) << 16) | ((q3 & 0xFF) << 24);
    }
}

// ============================================================
// Kernel 5: int8 WMMA GEMM
//   block = 256 threads = 8 wave32 arranged 2(M) x 4(N)
//   block tile 128x256, wave tile 64x64 (4x4 fragments of 16x16)
//   K-loop: 64 steps; per step 16 global_load_b128 + 16 WMMAs
// ============================================================
__device__ __forceinline__ v8i load_frag8(const char* __restrict__ p) {
    union { v8i v; uint4 q[2]; } f;
    f.q[0] = *(const uint4*)(p + 0);
    f.q[1] = *(const uint4*)(p + 16);
    return f.v;
}

__global__ __launch_bounds__(256) void bitlinear_gemm_kernel(
    const char* __restrict__ aq, const char* __restrict__ wq,
    const float* __restrict__ comb, float* __restrict__ out)
{
    const int lane = threadIdx.x & 31;
    const int wave = threadIdx.x >> 5;     // 0..7
    const int wm   = wave >> 2;            // 0..1  (M direction)
    const int wn   = wave & 3;             // 0..3  (N direction)
    const int l15  = lane & 15;
    const int half = lane >> 4;

    const int m_blk = blockIdx.y * 128;
    const int n_blk = blockIdx.x * 256;
    const int m0 = m_blk + wm * 64;
    const int n0 = n_blk + wn * 64;

    // stage the 128 per-row output scales for this block tile
    __shared__ float s_comb[128];
    if (threadIdx.x < 128) s_comb[threadIdx.x] = comb[m_blk + threadIdx.x];
    __syncthreads();

    const char* aptr[4];
    const char* bptr[4];
    #pragma unroll
    for (int i = 0; i < 4; ++i)
        aptr[i] = aq + (size_t)(m0 + i * 16 + l15) * K_IN + half * 32;
    #pragma unroll
    for (int j = 0; j < 4; ++j)
        bptr[j] = wq + (size_t)(n0 + j * 16 + l15) * K_IN + half * 32;

    v8i acc[4][4] = {};

    for (int kk = 0; kk < K_IN; kk += 64) {
        v8i afrag[4], bfrag[4];
        #pragma unroll
        for (int j = 0; j < 4; ++j) bfrag[j] = load_frag8(bptr[j] + kk);
        #pragma unroll
        for (int i = 0; i < 4; ++i) afrag[i] = load_frag8(aptr[i] + kk);
        #pragma unroll
        for (int i = 0; i < 4; ++i)
            #pragma unroll
            for (int j = 0; j < 4; ++j)
                acc[i][j] = __builtin_amdgcn_wmma_i32_16x16x64_iu8(
                    /*sgn_a=*/true, afrag[i],
                    /*sgn_b=*/true, bfrag[j],
                    acc[i][j], /*reuse_a=*/false, /*reuse_b=*/false);
    }

    // epilogue: C/D layout -> M = r + half*8, N = l15 within each 16x16 frag
    #pragma unroll
    for (int i = 0; i < 4; ++i) {
        #pragma unroll
        for (int j = 0; j < 4; ++j) {
            union { v8i v; int s[8]; } c;
            c.v = acc[i][j];
            const int col = n0 + j * 16 + l15;
            #pragma unroll
            for (int r = 0; r < 8; ++r) {
                const int lrow = wm * 64 + i * 16 + half * 8 + r; // row in block tile
                out[(size_t)(m_blk + lrow) * N_OUT + col] =
                    (float)c.s[r] * s_comb[lrow];
            }
        }
    }
}

// ============================================================
// launcher
// ============================================================
extern "C" void kernel_launch(void* const* d_in, const int* in_sizes, int n_in,
                              void* d_out, int out_size, void* d_ws, size_t ws_size,
                              hipStream_t stream) {
    const float* d_x = (const float*)d_in[0];   // [2,4096,4096] f32
    const float* d_w = (const float*)d_in[1];   // [4096,4096]   f32
    float* d_y = (float*)d_out;                 // [2,4096,4096] f32

    char* ws = (char*)d_ws;
    float* d_wscale  = (float*)ws;
    float* d_partial = (float*)(ws + WS_PARTIAL_OFF);
    float* d_comb    = (float*)(ws + WS_COMB_OFF);
    char*  d_wq      = ws + WS_WQ_OFF;
    char*  d_aq      = ws + WS_AQ_OFF;

    wabs_partial_kernel<<<1024, 256, 0, stream>>>((const float4*)d_w, d_partial);
    wscale_final_kernel<<<1, 256, 0, stream>>>(d_partial, d_wscale);
    quant_w_kernel<<<W_ELEMS / 4 / 256, 256, 0, stream>>>(
        (const float4*)d_w, d_wscale, (int*)d_wq);
    quant_a_kernel<<<M_TOK, 256, 0, stream>>>(d_x, d_wscale, (int*)d_aq, d_comb);

    dim3 grid(N_OUT / 256, M_TOK / 128);
    bitlinear_gemm_kernel<<<grid, 256, 0, stream>>>(d_aq, d_wq, d_comb, d_y);
}
